// GaussianSDEFinancialMultivariate_75737453298065
// MI455X (gfx1250) — compile-verified, matrix operand order checked by reading, not run on hardware
//
#include <hip/hip_runtime.h>
#include <cmath>

// ---------------------------------------------------------------------------
// GaussianSDE posterior simulation for MI455X (gfx1250, wave32, WMMA)
//   path[B,T,32], kl[B]  from  x0, time_sequence, noise, 3 MLP param sets
// Strategy: 32 batch rows / workgroup, T-loop in-kernel, activations in LDS,
// all GEMMs via v_wmma_f32_16x16x32_f16 (f16 in, f32 accumulate),
// weights pre-converted/padded to f16 in d_ws by a prep kernel.
// v2: 8 waves/WG, native v_tanh_f32, branch-free softplus, weight prefetch.
// v3: cholesky-MLP weights (180KB f16) staged ONCE into LDS via
//     global_load_async_to_lds_b128 (ASYNCcnt DMA) -> chol B-fragments become
//     ds_load_b128; cuts ~23% of per-step L2 weight traffic.
// ---------------------------------------------------------------------------

#define STATE_DIM 32
#define BATCH     4096
#define TSTEPS    128
#define BROWS     32          // batch rows per workgroup (2 WMMA row tiles)
#define NWAVES    8
#define NTHREADS  (NWAVES * 32)
#define N_TRIL    528         // 32*33/2
#define SOFTPLUS_ONE 0.5413248546129181f
#define SIGMA2    0.25f       // SIGMA=0.5

// LDS layout (halfs): sA 32*256 | sB 32*256 | sChol 32*528 | chol weights
#define WCH_H1    (128 * 64)          // 8192 halfs
#define WCH_H2    (128 * 128)         // 16384 halfs
#define WCH_H3    (N_TRIL * 128)      // 67584 halfs
#define WCH_TOTAL (WCH_H1 + WCH_H2 + WCH_H3)              // 92160 halfs
#define OFF_WCH_BYTES ((32 * 256 + 32 * 256 + 32 * N_TRIL) * 2)

typedef __attribute__((ext_vector_type(16))) _Float16 v16h;
typedef __attribute__((ext_vector_type(8)))  _Float16 v8h;
typedef __attribute__((ext_vector_type(8)))  float    v8f;

struct SdeParams {
  // padded f16 weights in workspace ([N][Kpad], row-major)
  const _Float16 *dW1, *dW2, *dW3, *dW4;   // drift 64->256,256->256,256->256,256->32
  const _Float16 *cW1, *cW2, *cW3, *cW4;   // ctrl  same shapes
  const _Float16 *hW1, *hW2, *hW3;         // chol  64->128,128->128,128->528 (contiguous!)
  // original f32 biases
  const float *db1, *db2, *db3, *db4;
  const float *cb1, *cb2, *cb3, *cb4;
  const float *hb1, *hb2, *hb3;
  const float *x0, *ts, *noise;
  float *path, *kl;
};

// Branch-free tanh: native v_tanh_f32 on gfx1250 (confirmed in round-2 asm).
__device__ __forceinline__ float fast_tanh(float x) {
#if __has_builtin(__builtin_amdgcn_tanhf)
  return __builtin_amdgcn_tanhf(x);
#elif __has_builtin(__builtin_amdgcn_tanh_f32)
  return __builtin_amdgcn_tanh_f32(x);
#else
  float t = __builtin_amdgcn_exp2f(2.885390081777927f * x);  // e^{2x}
  return 1.0f - 2.0f * __builtin_amdgcn_rcpf(t + 1.0f);
#endif
}

// Branch-free softplus: log(1+e^v) via v_exp_f32 + v_log_f32.
__device__ __forceinline__ float softplus_f(float v) {
  float e = __builtin_amdgcn_exp2f(1.4426950408889634f * v);  // e^v
#if __has_builtin(__builtin_amdgcn_logf)
  return 0.6931471805599453f * __builtin_amdgcn_logf(1.0f + e);
#else
  return logf(1.0f + e);
#endif
}

// Compose a v16h WMMA fragment from two contiguous 8-half (16B) loads,
// following the CDNA5 16-bit A/B layout:
//   lanes 0-15 : row = lane,    elems 0..7 = K[kc..kc+7],   8..15 = K[kc+16..kc+23]
//   lanes16-31 : row = lane-16, elems 0..7 = K[kc+8..kc+15],8..15 = K[kc+24..kc+31]
__device__ __forceinline__ v16h make_frag(const _Float16* base, int ld,
                                          int row, int kh) {
  v8h lo = *(const v8h*)(base + (size_t)row * ld + kh);
  v8h hi = *(const v8h*)(base + (size_t)row * ld + kh + 16);
  v16h r;
#pragma unroll
  for (int i = 0; i < 8; ++i) { r[i] = lo[i]; r[i + 8] = hi[i]; }
  return r;
}

// One dense layer over the 32-row block:  out = act(in[32xK] * W^T[KxN] + b)
// A (activations) from LDS; B (weights [N][K] f16) from global (L2-resident)
// or LDS (staged chol weights, PREF=false).
// NWAVES waves split N-tiles round-robin; each wave computes both 16-row tiles.
template<bool TANH, bool OUTF16, bool PREF>
__device__ __forceinline__ void mlp_layer(
    const _Float16* __restrict__ W, const float* __restrict__ bias,
    const _Float16* sIn, int K, void* sOut, int ldo, int N,
    int wave, int lane)
{
  const int NT   = N >> 4;
  const int l15  = lane & 15;
  const int khof = (lane >> 4) << 3;     // 0 or 8

  for (int nt = wave; nt < NT; nt += NWAVES) {
    const int n0 = nt << 4;
    // prefetch this wave's NEXT weight tile toward the WGP (global_prefetch_b8)
    if (PREF && nt + NWAVES < NT)
      __builtin_prefetch(W + (size_t)((nt + NWAVES) * 16 + l15) * K, 0, 1);

    const float bv = bias[n0 + l15];
    v8f acc0 = {bv, bv, bv, bv, bv, bv, bv, bv};
    v8f acc1 = acc0;
    for (int kc = 0; kc < K; kc += 32) {
      const int kh = kc + khof;
      v16h bfrag = make_frag(W,   K, n0 + l15, kh);   // weight row = out neuron
      v16h a0    = make_frag(sIn, K, l15,      kh);   // row tile 0
      v16h a1    = make_frag(sIn, K, l15 + 16, kh);   // row tile 1
      acc0 = __builtin_amdgcn_wmma_f32_16x16x32_f16(
                 false, a0, false, bfrag, (short)0, acc0, false, false);
      acc1 = __builtin_amdgcn_wmma_f32_16x16x32_f16(
                 false, a1, false, bfrag, (short)0, acc1, false, false);
    }
    // C/D layout: VGPR j -> M = j + 8*(lane>=16), N = n0 + (lane&15)
    const int n  = n0 + l15;
    const int mb = (lane >> 4) << 3;
#pragma unroll
    for (int j = 0; j < 8; ++j) {
      float v0 = acc0[j], v1 = acc1[j];
      if (TANH) { v0 = fast_tanh(v0); v1 = fast_tanh(v1); }
      if (OUTF16) {
        ((_Float16*)sOut)[(size_t)(mb + j)      * ldo + n] = (_Float16)v0;
        ((_Float16*)sOut)[(size_t)(mb + j + 16) * ldo + n] = (_Float16)v1;
      } else {
        ((float*)sOut)[(size_t)(mb + j)      * ldo + n] = v0;
        ((float*)sOut)[(size_t)(mb + j + 16) * ldo + n] = v1;
      }
    }
  }
}

// Convert f32 weight [dout][din] -> f16 padded [dout][K] (zero-fill K>=din).
__global__ void pad_convert_kernel(const float* __restrict__ src,
                                   _Float16* __restrict__ dst,
                                   int dout, int din, int K) {
  int idx = blockIdx.x * blockDim.x + threadIdx.x;
  if (idx >= dout * K) return;
  int n = idx / K, k = idx - n * K;
  dst[idx] = (k < din) ? (_Float16)src[(size_t)n * din + k] : (_Float16)0.0f;
}

__global__ __launch_bounds__(NTHREADS)
void sde_posterior_kernel(SdeParams p) {
  extern __shared__ char smem[];
  _Float16* sA    = (_Float16*)smem;           // [32][256]  activations ping
  _Float16* sB    = sA + 32 * 256;             // [32][256]  activations pong
  _Float16* sChol = sB + 32 * 256;             // [32][528]  packed tril output
  _Float16* sWch  = sChol + 32 * N_TRIL;       // staged chol weights (92160 halfs)
  float* sF    = (float*)(sWch + WCH_TOTAL);
  float* sX    = sF;                           // [32][32] state
  float* sDrift= sX + 1024;                    // [32][32]
  float* sU    = sDrift + 1024;                // [32][32]
  float* sLtu  = sU + 1024;                    // [32][32]
  float* sEps  = sLtu + 1024;                  // [32][32]
  float* sKL   = sEps + 1024;                  // [32]

  const int tid  = threadIdx.x;
  const int wave = tid >> 5;
  const int lane = tid & 31;
  const int row0 = blockIdx.x * BROWS;

  // ---- one-time async DMA: chol weights (contiguous in d_ws) -> LDS -------
  // global_load_async_to_lds_b128: VDST = LDS byte offset, VADDR = 64b addr;
  // tracked by ASYNCcnt, no VGPR data round-trip.
  {
    const char* gsrc = (const char*)p.hW1;
    const int nchunks = (WCH_TOTAL * 2) / 16;       // 16B per op
    for (int c = tid; c < nchunks; c += NTHREADS) {
      unsigned lds_off = (unsigned)(OFF_WCH_BYTES + c * 16);
      const char* gaddr = gsrc + (size_t)c * 16;
      asm volatile("global_load_async_to_lds_b128 %0, %1, off"
                   :: "v"(lds_off), "v"(gaddr) : "memory");
    }
    asm volatile("s_wait_asynccnt 0" ::: "memory");
  }

  // init state + kl
  for (int idx = tid; idx < 1024; idx += NTHREADS)
    sX[idx] = p.x0[(size_t)(row0 + (idx >> 5)) * STATE_DIM + (idx & 31)];
  if (tid < 32) sKL[tid] = 0.0f;
  __syncthreads();

  const _Float16* sWh1 = sWch;
  const _Float16* sWh2 = sWch + WCH_H1;
  const _Float16* sWh3 = sWch + WCH_H1 + WCH_H2;

  float tprev = p.ts[0];
  for (int it = 0; it < TSTEPS; ++it) {
    const float tcur = p.ts[it];
    const float dt   = tcur - tprev;
    const float sqdt = sqrtf(dt);
    tprev = tcur;

    // ================= drift MLP: [t,X] -> 256 -> 256 -> 256 -> 32 ==========
    for (int idx = tid; idx < 32 * 64; idx += NTHREADS) {
      int r = idx >> 6, c = idx & 63;
      float v = (c == 0) ? tcur : (c <= 32 ? sX[r * 32 + (c - 1)] : 0.0f);
      sA[idx] = (_Float16)v;
    }
    __syncthreads();
    mlp_layer<true,  true,  true>(p.dW1, p.db1, sA,  64, sB,     256, 256, wave, lane); __syncthreads();
    mlp_layer<true,  true,  true>(p.dW2, p.db2, sB, 256, sA,     256, 256, wave, lane); __syncthreads();
    mlp_layer<true,  true,  true>(p.dW3, p.db3, sA, 256, sB,     256, 256, wave, lane); __syncthreads();
    mlp_layer<false, false, true>(p.dW4, p.db4, sB, 256, sDrift,  32,  32, wave, lane); __syncthreads();

    // ================= ctrl MLP -> u =======================================
    for (int idx = tid; idx < 32 * 64; idx += NTHREADS) {
      int r = idx >> 6, c = idx & 63;
      float v = (c == 0) ? tcur : (c <= 32 ? sX[r * 32 + (c - 1)] : 0.0f);
      sA[idx] = (_Float16)v;
    }
    __syncthreads();
    mlp_layer<true,  true,  true>(p.cW1, p.cb1, sA,  64, sB,  256, 256, wave, lane); __syncthreads();
    mlp_layer<true,  true,  true>(p.cW2, p.cb2, sB, 256, sA,  256, 256, wave, lane); __syncthreads();
    mlp_layer<true,  true,  true>(p.cW3, p.cb3, sA, 256, sB,  256, 256, wave, lane); __syncthreads();
    mlp_layer<false, false, true>(p.cW4, p.cb4, sB, 256, sU,   32,  32, wave, lane); __syncthreads();

    // ======== cholesky MLP -> 528 tril (weights LDS-resident) ==============
    for (int idx = tid; idx < 32 * 64; idx += NTHREADS) {
      int r = idx >> 6, c = idx & 63;
      float v = (c == 0) ? tcur : (c <= 32 ? sX[r * 32 + (c - 1)] : 0.0f);
      sA[idx] = (_Float16)v;
    }
    __syncthreads();
    mlp_layer<true,  true,  false>(sWh1, p.hb1, sA,  64, sB,    128,    128, wave, lane); __syncthreads();
    mlp_layer<true,  true,  false>(sWh2, p.hb2, sB, 128, sA,    128,    128, wave, lane); __syncthreads();
    mlp_layer<false, true,  false>(sWh3, p.hb3, sA, 128, sChol, N_TRIL, N_TRIL, wave, lane); __syncthreads();

    // ================= combine: Lt_u, KL ===================================
    for (int idx = tid; idx < 1024; idx += NTHREADS) {
      int r = idx >> 5, i = idx & 31;
      sEps[idx] = p.noise[((size_t)(row0 + r) * TSTEPS + it) * STATE_DIM + i];
      float s = 0.0f;
      for (int j = i; j < STATE_DIM; ++j) {
        float lv = (float)sChol[r * N_TRIL + ((j * (j + 1)) >> 1) + i];
        if (j == i) lv = softplus_f(lv + SOFTPLUS_ONE);
        s += lv * sU[r * 32 + j];
      }
      sLtu[idx] = s;
      atomicAdd(&sKL[r], 0.5f * SIGMA2 * s * s * dt);   // ds_add_f32
    }
    __syncthreads();
    // ============ X += (drift - s2*L*Ltu)*dt + SIGMA*sqrt(dt)*L*eps ========
    for (int idx = tid; idx < 1024; idx += NTHREADS) {
      int r = idx >> 5, i = idx & 31;
      float a1 = 0.0f, a2 = 0.0f;
      for (int j = 0; j <= i; ++j) {
        float lv = (float)sChol[r * N_TRIL + ((i * (i + 1)) >> 1) + j];
        if (j == i) lv = softplus_f(lv + SOFTPLUS_ONE);
        a1 += lv * sLtu[r * 32 + j];
        a2 += lv * sEps[r * 32 + j];
      }
      float xn = sX[idx] + (sDrift[idx] - SIGMA2 * a1) * dt + 0.5f * sqdt * a2;
      sX[idx] = xn;
      p.path[((size_t)(row0 + r) * TSTEPS + it) * STATE_DIM + i] = xn;
    }
    __syncthreads();
  }
  if (tid < 32) p.kl[row0 + tid] = sKL[tid];
}

extern "C" void kernel_launch(void* const* d_in, const int* in_sizes, int n_in,
                              void* d_out, int out_size, void* d_ws, size_t ws_size,
                              hipStream_t stream) {
  (void)in_sizes; (void)n_in; (void)out_size; (void)ws_size;
  // setup_inputs order: x0, time_sequence, noise,
  //   drift (W1,b1,W2,b2,W3,b3,W4,b4), diff/chol (W1,b1,W2,b2,W3,b3),
  //   ctrl (W1,b1,W2,b2,W3,b3,W4,b4)
  const float* x0    = (const float*)d_in[0];
  const float* ts    = (const float*)d_in[1];
  const float* noise = (const float*)d_in[2];
  const float* dW1f  = (const float*)d_in[3];
  const float* db1   = (const float*)d_in[4];
  const float* dW2f  = (const float*)d_in[5];
  const float* db2   = (const float*)d_in[6];
  const float* dW3f  = (const float*)d_in[7];
  const float* db3   = (const float*)d_in[8];
  const float* dW4f  = (const float*)d_in[9];
  const float* db4   = (const float*)d_in[10];
  const float* hW1f  = (const float*)d_in[11];
  const float* hb1   = (const float*)d_in[12];
  const float* hW2f  = (const float*)d_in[13];
  const float* hb2   = (const float*)d_in[14];
  const float* hW3f  = (const float*)d_in[15];
  const float* hb3   = (const float*)d_in[16];
  const float* cW1f  = (const float*)d_in[17];
  const float* cb1   = (const float*)d_in[18];
  const float* cW2f  = (const float*)d_in[19];
  const float* cb2   = (const float*)d_in[20];
  const float* cW3f  = (const float*)d_in[21];
  const float* cb3   = (const float*)d_in[22];
  const float* cW4f  = (const float*)d_in[23];
  const float* cb4   = (const float*)d_in[24];

  _Float16* ws = (_Float16*)d_ws;
  size_t off = 0;
  auto take = [&](size_t n) { _Float16* q = ws + off; off += n; return q; };
  _Float16* dW1p = take(256 * 64);
  _Float16* dW2p = take(256 * 256);
  _Float16* dW3p = take(256 * 256);
  _Float16* dW4p = take(32  * 256);
  _Float16* cW1p = take(256 * 64);
  _Float16* cW2p = take(256 * 256);
  _Float16* cW3p = take(256 * 256);
  _Float16* cW4p = take(32  * 256);
  _Float16* hW1p = take(WCH_H1);   // chol weights contiguous: hW1|hW2|hW3
  _Float16* hW2p = take(WCH_H2);
  _Float16* hW3p = take(WCH_H3);

  auto conv = [&](const float* src, _Float16* dst, int dout, int din, int K) {
    int total = dout * K;
    pad_convert_kernel<<<(total + 255) / 256, 256, 0, stream>>>(src, dst, dout, din, K);
  };
  conv(dW1f, dW1p, 256, 33, 64);
  conv(dW2f, dW2p, 256, 256, 256);
  conv(dW3f, dW3p, 256, 256, 256);
  conv(dW4f, dW4p, 32, 256, 256);
  conv(cW1f, cW1p, 256, 33, 64);
  conv(cW2f, cW2p, 256, 256, 256);
  conv(cW3f, cW3p, 256, 256, 256);
  conv(cW4f, cW4p, 32, 256, 256);
  conv(hW1f, hW1p, 128, 33, 64);
  conv(hW2f, hW2p, 128, 128, 128);
  conv(hW3f, hW3p, N_TRIL, 128, 128);

  float* path = (float*)d_out;
  float* kl   = path + (size_t)BATCH * TSTEPS * STATE_DIM;

  SdeParams p;
  p.dW1 = dW1p; p.dW2 = dW2p; p.dW3 = dW3p; p.dW4 = dW4p;
  p.cW1 = cW1p; p.cW2 = cW2p; p.cW3 = cW3p; p.cW4 = cW4p;
  p.hW1 = hW1p; p.hW2 = hW2p; p.hW3 = hW3p;
  p.db1 = db1; p.db2 = db2; p.db3 = db3; p.db4 = db4;
  p.cb1 = cb1; p.cb2 = cb2; p.cb3 = cb3; p.cb4 = cb4;
  p.hb1 = hb1; p.hb2 = hb2; p.hb3 = hb3;
  p.x0 = x0; p.ts = ts; p.noise = noise;
  p.path = path; p.kl = kl;

  const size_t smem_bytes =
      (size_t)(32 * 256 + 32 * 256 + 32 * N_TRIL + WCH_TOTAL) * sizeof(_Float16) +
      (size_t)(5 * 1024 + 32) * sizeof(float);   // ~271 KB (< 320 KB/WG)

  sde_posterior_kernel<<<BATCH / BROWS, NTHREADS, smem_bytes, stream>>>(p);
}
